// Attention_85993835200662
// MI455X (gfx1250) — compile-verified, matrix operand order checked by reading, not run on hardware
//
#include <hip/hip_runtime.h>

// ---------------------------------------------------------------------------
// Fused multi-head attention with cross-head renormalization for gfx1250.
//   b=4, n=2048, d=512, h=8, hd=64.
//   All matmuls via v_wmma_f32_16x16x32_bf16 (fp32 accumulate).
//   Projection GEMMs stage their shared A panel in LDS with
//   global_load_async_to_lds_b128 (ASYNCcnt). Attention is 2-pass
//   (softmax stats, then fused cross-head renorm + P@V) so the 537MB
//   attention matrix never touches HBM.
// ---------------------------------------------------------------------------

typedef __attribute__((ext_vector_type(16))) __bf16 v16bf;
typedef __attribute__((ext_vector_type(8)))  float  v8f;

#define FEAT   512
#define NHEAD  8
#define HD     64
#define SEQ    2048
#define BATCH  4
#define NTOK   (BATCH * SEQ)      // 8192
#define SCALE  8.0f               // sqrt(head_dim)
#define EPS_A  1e-6f

static __device__ __forceinline__ v8f wmma_bf16(v16bf a, v16bf b, v8f c) {
  // 8 args: (neg_a, A, neg_b, B, c_mod, C, reuse_a, reuse_b)
  return __builtin_amdgcn_wmma_f32_16x16x32_bf16(false, a, false, b, (short)0, c,
                                                 false, false);
}

static __device__ __forceinline__ __bf16 to_bf16(float f) { return (__bf16)f; }

// ds_swizzle_b32 XOR within 16 lanes: offset15=0, and=0x1f, or=0, xor=mask
#define SWZ_XOR(v, imm) \
  __int_as_float(__builtin_amdgcn_ds_swizzle(__float_as_int(v), imm))

// ---------------------------------------------------------------------------
// fp32 -> bf16 conversion (grid-stride)
// ---------------------------------------------------------------------------
__global__ void cvt_f32_bf16(const float* __restrict__ src,
                             __bf16* __restrict__ dst, int n) {
  int i = blockIdx.x * blockDim.x + threadIdx.x;
  int stride = gridDim.x * blockDim.x;
  for (; i < n; i += stride) dst[i] = to_bf16(src[i]);
}

// ---------------------------------------------------------------------------
// C = A (M x 512 bf16, row-major) * B(512 x 512 bf16 "W", row-major)^T
// One block = one 16-row panel of A (shared by 8 waves via async copy to
// LDS); wave w computes the 16x64 tile at columns w*64..w*64+63.
// MODE 0: scatter bf16 to (b, h, n, hd)      [Q, K]
// MODE 1: scatter bf16 to (b, h, hd, n)      [V transposed for P@V B-frags]
// MODE 2: row-major fp32 (M x 512)           [final output]
// ---------------------------------------------------------------------------
template <int MODE>
__global__ __launch_bounds__(256) void gemm_proj(const __bf16* __restrict__ A,
                                                 const __bf16* __restrict__ B,
                                                 void* __restrict__ out, int M) {
  __shared__ __align__(128) __bf16 Alds[16 * FEAT];  // 16 KB panel

  const int tid  = threadIdx.x;
  const int lane = tid & 31;
  const int wave = tid >> 5;
  const int lp   = lane & 15;
  const int half = lane >> 4;

  const int tm = blockIdx.x;   // row tile (uniform across block)
  const int tn = wave;         // column tile = wave id (tilesN == waves == 8)
  if (tm * 16 >= M) return;    // uniform exit

  // --- cooperative async copy of the 16x512 A panel into LDS -------------
  {
    const char* src = (const char*)(A + (long)tm * 16 * FEAT);  // contiguous 16KB
    const unsigned dstBase = (unsigned)(size_t)&Alds[0];
#pragma unroll
    for (int s = 0; s < 4; ++s) {
      const unsigned off = (unsigned)(tid + s * 256) * 16u;  // 16B per lane
      const unsigned ldsAddr = dstBase + off;
      const char* g = src + off;
      asm volatile("global_load_async_to_lds_b128 %0, %1, off"
                   :: "v"(ldsAddr), "v"(g)
                   : "memory");
    }
    asm volatile("s_wait_asynccnt 0" ::: "memory");
    __syncthreads();
  }

  v8f acc[4] = {};
  for (int k0 = 0; k0 < FEAT; k0 += 32) {
    v16bf a = *(const v16bf*)&Alds[lp * FEAT + k0 + half * 16];
    if (k0 + 64 < FEAT)  // keep B stream ahead (global_prefetch_b8)
      __builtin_prefetch(&B[((long)tn * 64 + lp) * FEAT + k0 + 64], 0, 0);
#pragma unroll
    for (int f = 0; f < 4; ++f) {
      const int col = tn * 64 + f * 16 + lp;
      v16bf bf = *(const v16bf*)&B[(long)col * FEAT + k0 + half * 16];
      acc[f] = wmma_bf16(a, bf, acc[f]);
    }
  }

#pragma unroll
  for (int f = 0; f < 4; ++f) {
    const int col = tn * 64 + f * 16 + lp;
    const int h = col >> 6, d = col & 63;
#pragma unroll
    for (int i = 0; i < 8; ++i) {
      const int r  = tm * 16 + i + 8 * half;  // C layout: M = vgpr + 8*(lane>=16)
      const int bb = r / SEQ, rr = r % SEQ;
      const float v = acc[f][i];
      if (MODE == 0)
        ((__bf16*)out)[(((long)(bb * NHEAD + h) * SEQ) + rr) * HD + d] = to_bf16(v);
      else if (MODE == 1)
        ((__bf16*)out)[(((long)(bb * NHEAD + h) * HD) + d) * SEQ + rr] = to_bf16(v);
      else
        ((float*)out)[(long)r * FEAT + col] = v;
    }
  }
}

// ---------------------------------------------------------------------------
// Pass 1: per-(b,h,row) softmax max + denom over all 2048 keys.
// One wave per 16-query tile. Deferred reduction: each lane keeps a running
// max/sum over its own key columns; a single cross-lane combine (ds_swizzle)
// happens after the key loop.
// ---------------------------------------------------------------------------
__global__ __launch_bounds__(256) void softmax_stats(
    const __bf16* __restrict__ Qh, const __bf16* __restrict__ Kh,
    float* __restrict__ Mrow, float* __restrict__ Lrow) {
  const int tid  = threadIdx.x;
  const int lane = tid & 31;
  const int wave = tid >> 5;
  const int lp   = lane & 15;
  const int half = lane >> 4;

  const int t  = blockIdx.x * (blockDim.x >> 5) + wave;  // 0..4095
  const int bh = t >> 7;                                 // (b*8+h)
  const int qt = t & 127;

  const __bf16* qb = Qh + ((long)bh * SEQ + qt * 16 + lp) * HD;
  v16bf a0 = *(const v16bf*)&qb[half * 16];
  v16bf a1 = *(const v16bf*)&qb[32 + half * 16];

  float m[8], l[8];  // per-LANE running stats (this lane's key columns only)
#pragma unroll
  for (int i = 0; i < 8; ++i) { m[i] = -3.0e38f; l[i] = 0.f; }

  for (int kt = 0; kt < SEQ / 16; ++kt) {
    const __bf16* kb = Kh + ((long)bh * SEQ + kt * 16 + lp) * HD;
    v16bf b0 = *(const v16bf*)&kb[half * 16];
    v16bf b1 = *(const v16bf*)&kb[32 + half * 16];
    v8f c = {};
    c = wmma_bf16(a0, b0, c);
    c = wmma_bf16(a1, b1, c);
#pragma unroll
    for (int i = 0; i < 8; ++i) {
      const float s  = c[i] * SCALE;
      const float mn = fmaxf(m[i], s);
      l[i] = l[i] * __expf(m[i] - mn) + __expf(s - mn);
      m[i] = mn;
    }
  }

  // single cross-lane combine over the 16 lanes of each half-wave
#pragma unroll
  for (int i = 0; i < 8; ++i) {
    float M = m[i];
    M = fmaxf(M, SWZ_XOR(M, 0x041F));
    M = fmaxf(M, SWZ_XOR(M, 0x081F));
    M = fmaxf(M, SWZ_XOR(M, 0x101F));
    M = fmaxf(M, SWZ_XOR(M, 0x201F));
    float L = l[i] * __expf(m[i] - M);
    L += SWZ_XOR(L, 0x041F);
    L += SWZ_XOR(L, 0x081F);
    L += SWZ_XOR(L, 0x101F);
    L += SWZ_XOR(L, 0x201F);
    if (lp == 0) {
      const int r = qt * 16 + i + 8 * half;
      Mrow[(long)bh * SEQ + r] = M;
      Lrow[(long)bh * SEQ + r] = L;
    }
  }
}

// ---------------------------------------------------------------------------
// Pass 2: one workgroup per (b, 16-query tile); 8 waves = 8 heads.
// Per 32-key block: scores -> probs (kept in registers + published to LDS)
// -> cooperative reciprocal of cross-head totals -> w = p*rtot + eps (bf16)
// -> WMMA against transposed V.
// ---------------------------------------------------------------------------
__global__ __launch_bounds__(256) void attn_pass2(
    const __bf16* __restrict__ Qh, const __bf16* __restrict__ Kh,
    const __bf16* __restrict__ Vt, const float* __restrict__ Mrow,
    const float* __restrict__ Lrow, __bf16* __restrict__ O) {
  __shared__ __align__(64) float  ptile[NHEAD][16][32];  // 16 KB
  __shared__ __align__(64) float  rtot[16][32];          //  2 KB
  __shared__ __align__(64) __bf16 wtile[NHEAD][16][32];  //  8 KB

  const int tid  = threadIdx.x;
  const int lane = tid & 31;
  const int h    = tid >> 5;      // wave == head
  const int lp   = lane & 15;
  const int half = lane >> 4;

  const int b  = blockIdx.x >> 7;
  const int qt = blockIdx.x & 127;
  const int bh = b * NHEAD + h;

  const __bf16* qb = Qh + ((long)bh * SEQ + qt * 16 + lp) * HD;
  v16bf a0 = *(const v16bf*)&qb[half * 16];
  v16bf a1 = *(const v16bf*)&qb[32 + half * 16];

  float m[8], rl[8];
#pragma unroll
  for (int i = 0; i < 8; ++i) {
    const int r = qt * 16 + i + 8 * half;
    m[i]  = Mrow[(long)bh * SEQ + r];
    rl[i] = __frcp_rn(Lrow[(long)bh * SEQ + r]);
  }

  v8f o[4] = {};

  for (int key0 = 0; key0 < SEQ; key0 += 32) {
    float pk[2][8];  // this head's probs, kept in registers
#pragma unroll
    for (int t = 0; t < 2; ++t) {  // two 16-key score tiles
      const __bf16* kb = Kh + ((long)bh * SEQ + key0 + t * 16 + lp) * HD;
      v16bf b0 = *(const v16bf*)&kb[half * 16];
      v16bf b1 = *(const v16bf*)&kb[32 + half * 16];
      v8f c = {};
      c = wmma_bf16(a0, b0, c);
      c = wmma_bf16(a1, b1, c);
#pragma unroll
      for (int i = 0; i < 8; ++i) {
        const float p = __expf(c[i] * SCALE - m[i]) * rl[i];
        pk[t][i] = p;
        ptile[h][i + 8 * half][t * 16 + lp] = p;
      }
    }
    __syncthreads();  // all heads' probs visible

    // cooperative reciprocal cross-head totals: 512 elems, 2 per thread
#pragma unroll
    for (int e = 0; e < 2; ++e) {
      const int idx = tid * 2 + e;
      const int r = idx >> 5, c = idx & 31;
      float tot = 0.f;
#pragma unroll
      for (int hh = 0; hh < NHEAD; ++hh) tot += ptile[hh][r][c];
      rtot[r][c] = __frcp_rn(tot);
    }
    __syncthreads();

#pragma unroll
    for (int t = 0; t < 2; ++t)
#pragma unroll
      for (int i = 0; i < 8; ++i) {
        const int r = i + 8 * half, cidx = t * 16 + lp;
        const float w = pk[t][i] * rtot[r][cidx] + EPS_A;  // eps AFTER division
        wtile[h][r][cidx] = to_bf16(w);
      }

    // wtile[h] is written entirely by this wave; same-wave LDS ops are
    // in-order, so no workgroup barrier needed before the fragment read.
    v16bf wa = *(const v16bf*)&wtile[h][lp][half * 16];
#pragma unroll
    for (int f = 0; f < 4; ++f) {
      const __bf16* vb =
          Vt + ((long)bh * HD + f * 16 + lp) * SEQ + key0 + half * 16;
      v16bf bv = *(const v16bf*)vb;
      o[f] = wmma_bf16(wa, bv, o[f]);
    }
    __syncthreads();  // protect ptile/rtot against next iteration's writes
  }

#pragma unroll
  for (int f = 0; f < 4; ++f)
#pragma unroll
    for (int i = 0; i < 8; ++i) {
      const int r   = qt * 16 + i + 8 * half;
      const int col = h * HD + f * 16 + lp;
      O[((long)b * SEQ + r) * FEAT + col] = to_bf16(o[f][i]);
    }
}

// ---------------------------------------------------------------------------
// Host-side orchestration
// ---------------------------------------------------------------------------
extern "C" void kernel_launch(void* const* d_in, const int* in_sizes, int n_in,
                              void* d_out, int out_size, void* d_ws,
                              size_t ws_size, hipStream_t stream) {
  const float* x  = (const float*)d_in[0];
  const float* Wq = (const float*)d_in[1];
  const float* Wk = (const float*)d_in[2];
  const float* Wv = (const float*)d_in[3];
  const float* Wo = (const float*)d_in[4];

  char* ws = (char*)d_ws;
  size_t off = 0;
  auto carve = [&](size_t bytes) -> void* {
    off = (off + 255) & ~(size_t)255;
    void* p = ws + off;
    off += bytes;
    return p;
  };

  const size_t tokBytes = (size_t)NTOK * FEAT * sizeof(__bf16);  // 8 MB
  const size_t wBytes   = (size_t)FEAT * FEAT * sizeof(__bf16);
  const size_t hBytes   = (size_t)BATCH * NHEAD * SEQ * HD * sizeof(__bf16);

  __bf16* xb   = (__bf16*)carve(tokBytes);
  __bf16* wqb  = (__bf16*)carve(wBytes);
  __bf16* wkb  = (__bf16*)carve(wBytes);
  __bf16* wvb  = (__bf16*)carve(wBytes);
  __bf16* wob  = (__bf16*)carve(wBytes);
  __bf16* Qh   = (__bf16*)carve(hBytes);       // (b, h, n, hd)
  __bf16* Kh   = (__bf16*)carve(hBytes);       // (b, h, n, hd)
  __bf16* Vt   = (__bf16*)carve(hBytes);       // (b, h, hd, n)
  float*  Mrow = (float*)carve((size_t)BATCH * NHEAD * SEQ * sizeof(float));
  float*  Lrow = (float*)carve((size_t)BATCH * NHEAD * SEQ * sizeof(float));
  __bf16* Obuf = (__bf16*)carve(tokBytes);     // (b, n, h*hd)

  // 1) fp32 -> bf16
  cvt_f32_bf16<<<2048, 256, 0, stream>>>(x, xb, NTOK * FEAT);
  cvt_f32_bf16<<<256, 256, 0, stream>>>(Wq, wqb, FEAT * FEAT);
  cvt_f32_bf16<<<256, 256, 0, stream>>>(Wk, wkb, FEAT * FEAT);
  cvt_f32_bf16<<<256, 256, 0, stream>>>(Wv, wvb, FEAT * FEAT);
  cvt_f32_bf16<<<256, 256, 0, stream>>>(Wo, wob, FEAT * FEAT);

  // 2) Q/K/V projections: 512 blocks, each = one 16-row A panel x 8 col tiles
  gemm_proj<0><<<NTOK / 16, 256, 0, stream>>>(xb, wqb, Qh, NTOK);
  gemm_proj<0><<<NTOK / 16, 256, 0, stream>>>(xb, wkb, Kh, NTOK);
  gemm_proj<1><<<NTOK / 16, 256, 0, stream>>>(xb, wvb, Vt, NTOK);

  // 3) softmax stats: 4096 query tiles, 8 waves/block
  softmax_stats<<<512, 256, 0, stream>>>(Qh, Kh, Mrow, Lrow);

  // 4) fused cross-head renorm + P@V: (b * 128 qtiles) blocks, wave==head
  attn_pass2<<<BATCH * 128, 256, 0, stream>>>(Qh, Kh, Vt, Mrow, Lrow, Obuf);

  // 5) output projection -> fp32 d_out
  gemm_proj<2><<<NTOK / 16, 256, 0, stream>>>(Obuf, wob, d_out, NTOK);
}
